// SlidingWindowAttention_37933151158947
// MI455X (gfx1250) — matrix operand order repaired and checked by reading, not compiled, not run on hardware
//
#include <hip/hip_runtime.h>

// ---------------------------------------------------------------------------
// Sliding-window attention for MI455X (gfx1250), wave32 + WMMA bf16.
// B=2, S=4000, HID=1024, H=16, D=64, W=512, STRIDE=256, n_win=14 (+tail).
// All matmuls via v_wmma_f32_16x16x32_bf16 (fp32 accumulate).
// GEMMs: register-staged double-buffered LDS pipelining; fragment loads are
// batched ahead of the WMMA chain so each chain has a single ds wait.
// ---------------------------------------------------------------------------

typedef __bf16 bf16_t;
typedef __attribute__((ext_vector_type(16))) __bf16 v16bf;
typedef __attribute__((ext_vector_type(8)))  float  v8f;
typedef unsigned int u32x4 __attribute__((ext_vector_type(4)));

union AFrag { v16bf v; u32x4 u[2]; };
union CFrag { v8f v; float f[8]; };
union Pack16 { u32x4 u[2]; unsigned short s[16]; };

#define Bsz 2
#define Ssz 4000
#define HID 1024
#define NH 16
#define DH 64
#define WIN 512
#define NWIN 15   /* 14 sliding + 1 tail */
#define MROWS 8000 /* B*S */

__device__ __forceinline__ unsigned short f2bf(float f) {
  unsigned int u = __float_as_uint(f);
  u += 0x7FFFu + ((u >> 16) & 1u);   // round-to-nearest-even
  return (unsigned short)(u >> 16);
}

__device__ __forceinline__ v8f wmma_bf16(v16bf a, v16bf b, v8f c) {
  return __builtin_amdgcn_wmma_f32_16x16x32_bf16(false, a, false, b, (short)0, c,
                                                 false, false);
}

// A-matrix fragment (16x32 bf16): lane (half,ln) reads row (row0+ln),
// cols kbase + half*8 + [0..7] and kbase + 16 + half*8 + [0..7].
__device__ __forceinline__ v16bf load_a_frag(const unsigned short* base, int row0,
                                             int kbase, int pitch, int half, int ln) {
  AFrag f;
  const unsigned short* p = base + (row0 + ln) * pitch + kbase + half * 8;
  f.u[0] = *(const u32x4*)p;
  f.u[1] = *(const u32x4*)(p + 16);
  return f.v;
}

// B-matrix fragment (32x16 bf16), storage is [n][k] row-major:
// lane (half,ln) holds column n = n0+ln, K = kbase + half*8 + [0..7] and kbase + 16 + half*8 + [0..7].
__device__ __forceinline__ v16bf load_b_frag(const unsigned short* base, int n0,
                                             int kbase, int pitch, int half, int ln) {
  AFrag f;
  const unsigned short* p = base + (n0 + ln) * pitch + kbase + half * 8;
  f.u[0] = *(const u32x4*)p;
  f.u[1] = *(const u32x4*)(p + 16);
  return f.v;
}

// --------------------------- small utility kernels --------------------------

__global__ void swa_cvt_bf16_kernel(const float* __restrict__ src,
                                    unsigned short* __restrict__ dst, int n) {
  int i = blockIdx.x * blockDim.x + threadIdx.x;
  if (i < n) dst[i] = f2bf(src[i]);
}

// Wt[n][k] = bf16(W[k][n]) for 1024x1024
__global__ void swa_wtrans_kernel(const float* __restrict__ W,
                                  unsigned short* __restrict__ Wt) {
  int idx = blockIdx.x * blockDim.x + threadIdx.x;
  if (idx < HID * HID) {
    int n = idx & (HID - 1);
    int k = idx >> 10;
    Wt[n * HID + k] = f2bf(W[idx]);
  }
}

__global__ void swa_zero_kernel(float* __restrict__ p, int n) {
  int i = blockIdx.x * blockDim.x + threadIdx.x;
  if (i < n) p[i] = 0.0f;
}

// number of windows covering sequence position s (incl. tail), clipped >= 1
__device__ __forceinline__ float inv_count(int s) {
  int kmax = s >> 8; if (kmax > 13) kmax = 13;
  int t = s - (WIN - 1);
  int kmin = (t <= 0) ? 0 : ((t + 255) >> 8);
  int cnt = (kmin <= kmax) ? (kmax - kmin + 1) : 0;
  if (s >= Ssz - WIN) cnt += 1;
  if (cnt < 1) cnt = 1;
  return 1.0f / (float)cnt;
}

// ctx[b][s][h*64+d] = bf16( merged[b][h][s][d] / count(s) )
__global__ void swa_norm_kernel(const float* __restrict__ merged,
                                unsigned short* __restrict__ ctx) {
  int i = blockIdx.x * blockDim.x + threadIdx.x;
  if (i >= Bsz * Ssz * HID) return;
  int hid = i & (HID - 1);
  int s = (i >> 10) % Ssz;
  int b = (i >> 10) / Ssz;
  int h = hid >> 6, d = hid & 63;
  float v = merged[(((size_t)b * NH + h) * Ssz + s) * DH + d];
  ctx[i] = f2bf(v * inv_count(s));
}

// ------------------------------- GEMM core ----------------------------------
// Block tile 128(M) x 128(N), K-step 32 (one WMMA depth), double-buffered LDS.
// 8 waves in a 2x4 grid; each wave owns a 64x32 strip = 4x2 C tiles of 16x16.

#define GPITCH 40
#define GBUFE (128 * GPITCH)

__device__ __forceinline__ void gemm_tile(const unsigned short* __restrict__ A,
                                          const unsigned short* __restrict__ Bt,
                                          int Mtotal,
                                          unsigned short* sA, unsigned short* sB,
                                          CFrag acc[4][2]) {
  const int tid = threadIdx.x;
  const int lane = tid & 31, wave = tid >> 5;
  const int wr = wave & 1, wc = wave >> 1;
  const int half = lane >> 4, ln = lane & 15;
  const int lr = tid >> 1, lc = (tid & 1) * 16;

  int gm = blockIdx.x * 128 + lr; if (gm >= Mtotal) gm = Mtotal - 1;
  int gn = blockIdx.y * 128 + lr;
  const unsigned short* gA = A + (size_t)gm * HID + lc;
  const unsigned short* gB = Bt + (size_t)gn * HID + lc;

  for (int i = 0; i < 4; i++)
    for (int j = 0; j < 2; j++)
      for (int r = 0; r < 8; r++) acc[i][j].f[r] = 0.0f;

  // prologue: stage K-tile 0 through registers into LDS buffer 0
  u32x4 ra0 = *(const u32x4*)(gA);
  u32x4 ra1 = *(const u32x4*)(gA + 8);
  u32x4 rb0 = *(const u32x4*)(gB);
  u32x4 rb1 = *(const u32x4*)(gB + 8);
  int buf = 0;
  *(u32x4*)(sA + lr * GPITCH + lc)     = ra0;
  *(u32x4*)(sA + lr * GPITCH + lc + 8) = ra1;
  *(u32x4*)(sB + lr * GPITCH + lc)     = rb0;
  *(u32x4*)(sB + lr * GPITCH + lc + 8) = rb1;
  __syncthreads();

  for (int kb = 0; kb < HID; kb += 32) {
    const int nkb = kb + 32;
    if (nkb < HID) {   // issue next tile's loads early (overlap with WMMA)
      ra0 = *(const u32x4*)(gA + nkb);
      ra1 = *(const u32x4*)(gA + nkb + 8);
      rb0 = *(const u32x4*)(gB + nkb);
      rb1 = *(const u32x4*)(gB + nkb + 8);
      if (nkb + 32 < HID) {                      // global_prefetch_b8
        __builtin_prefetch(gA + nkb + 32, 0, 1);
        __builtin_prefetch(gB + nkb + 32, 0, 1);
      }
    }

    const unsigned short* cA = sA + buf * GBUFE;
    const unsigned short* cB = sB + buf * GBUFE;
    // batch ALL fragment ds_loads first, then run the WMMA chain
    v16bf bfr[2], af[4];
    for (int j = 0; j < 2; j++)
      bfr[j] = load_b_frag(cB, wc * 32 + j * 16, 0, GPITCH, half, ln);
    for (int i = 0; i < 4; i++)
      af[i] = load_a_frag(cA, wr * 64 + i * 16, 0, GPITCH, half, ln);
    for (int i = 0; i < 4; i++)
      for (int j = 0; j < 2; j++)
        acc[i][j].v = wmma_bf16(af[i], bfr[j], acc[i][j].v);

    if (nkb < HID) {   // store staged tile into the other buffer
      buf ^= 1;
      *(u32x4*)(sA + buf * GBUFE + lr * GPITCH + lc)     = ra0;
      *(u32x4*)(sA + buf * GBUFE + lr * GPITCH + lc + 8) = ra1;
      *(u32x4*)(sB + buf * GBUFE + lr * GPITCH + lc)     = rb0;
      *(u32x4*)(sB + buf * GBUFE + lr * GPITCH + lc + 8) = rb1;
      __syncthreads();
    }
  }
}

// QKV projection: A = X bf16 [8000][1024], Bt = [Wq^T;Wk^T;Wv^T] bf16 [3072][1024].
// Epilogue: +bias, convert bf16, scatter into split-head [B,H,S,D].
__global__ __launch_bounds__(256) void swa_gemm_qkv_kernel(
    const unsigned short* __restrict__ Xbf, const unsigned short* __restrict__ Wt3,
    const float* __restrict__ bq, const float* __restrict__ bk,
    const float* __restrict__ bv,
    unsigned short* __restrict__ Qo, unsigned short* __restrict__ Ko,
    unsigned short* __restrict__ Vo, int Mtotal) {
  __shared__ unsigned short sA[2 * GBUFE];
  __shared__ unsigned short sB[2 * GBUFE];
  CFrag acc[4][2];
  gemm_tile(Xbf, Wt3, Mtotal, sA, sB, acc);

  const int tid = threadIdx.x, lane = tid & 31, wave = tid >> 5;
  const int wr = wave & 1, wc = wave >> 1, half = lane >> 4, ln = lane & 15;
  for (int j = 0; j < 2; j++) {
    int ng = blockIdx.y * 128 + wc * 32 + j * 16 + ln;   // 0..3071
    int which = ng >> 10, nn = ng & (HID - 1);
    const float* bias = (which == 0) ? bq : (which == 1) ? bk : bv;
    unsigned short* dst = (which == 0) ? Qo : (which == 1) ? Ko : Vo;
    float bb_ = bias[nn];
    int h = nn >> 6, d = nn & 63;
    for (int i = 0; i < 4; i++) {
      for (int r = 0; r < 8; r++) {
        int ml = wr * 64 + i * 16 + r + 8 * half;
        int gm = blockIdx.x * 128 + ml;
        if (gm < Mtotal) {
          int b = gm / Ssz, s = gm - b * Ssz;
          dst[(((size_t)b * NH + h) * Ssz + s) * DH + d] =
              f2bf(acc[i][j].f[r] + bb_);
        }
      }
    }
  }
}

// Output projection: A = ctx bf16 [8000][1024], Bt = Wo^T, +bo, f32 out.
__global__ __launch_bounds__(256) void swa_gemm_out_kernel(
    const unsigned short* __restrict__ Cbf, const unsigned short* __restrict__ Wto,
    const float* __restrict__ bo, float* __restrict__ out, int Mtotal) {
  __shared__ unsigned short sA[2 * GBUFE];
  __shared__ unsigned short sB[2 * GBUFE];
  CFrag acc[4][2];
  gemm_tile(Cbf, Wto, Mtotal, sA, sB, acc);

  const int tid = threadIdx.x, lane = tid & 31, wave = tid >> 5;
  const int wr = wave & 1, wc = wave >> 1, half = lane >> 4, ln = lane & 15;
  for (int j = 0; j < 2; j++) {
    int ng = blockIdx.y * 128 + wc * 32 + j * 16 + ln;
    float bb_ = bo[ng];
    for (int i = 0; i < 4; i++) {
      for (int r = 0; r < 8; r++) {
        int ml = wr * 64 + i * 16 + r + 8 * half;
        int gm = blockIdx.x * 128 + ml;
        if (gm < Mtotal) out[(size_t)gm * HID + ng] = acc[i][j].f[r] + bb_;
      }
    }
  }
}

// ------------------------------ attention -----------------------------------
// One block = (b, h, win, 64-query tile). Flash-style over 8 chunks of 64 keys.
// Wave w: qr = w&3 (16-query strip), kg = w>>2 (32-wide key/d strip).
__global__ __launch_bounds__(256) void swa_attn_kernel(
    const unsigned short* __restrict__ Qbf, const unsigned short* __restrict__ Kbf,
    const unsigned short* __restrict__ Vbf, const float* __restrict__ pos_bias,
    float* __restrict__ merged) {
  __shared__ unsigned short Qs[64 * 72];
  __shared__ unsigned short Ks[64 * 72];
  __shared__ unsigned short Vt[64 * 72];   // transposed: [d][key]
  __shared__ unsigned short Ps[64 * 72];   // probs bf16
  __shared__ float Ss[64 * 68];            // scores f32
  __shared__ float red[64 * 4];            // per-row partial reductions
  __shared__ float mrow[64], lrow[64], rfac[64];

  int bid = blockIdx.x;
  const int qtile = bid & 7; bid >>= 3;
  const int win = bid % NWIN; bid /= NWIN;
  const int h = bid & (NH - 1);
  const int b = bid >> 4;
  const int start = (win < 14) ? win * 256 : (Ssz - WIN);

  const int tid = threadIdx.x, lane = tid & 31, wave = tid >> 5;
  const int half = lane >> 4, ln = lane & 15;
  const int qr = wave & 3;
  const int kg = wave >> 2;
  const float scale = 0.125f;  // 1/sqrt(64)

  const size_t headBase = (((size_t)b * NH + h) * Ssz) * DH;
  const int q0 = start + qtile * 64;
  const int rr = tid >> 2, seg = tid & 3;  // softmax: 4 threads per row

  { // load 64x64 Q tile
    int r = tid >> 2, c0 = (tid & 3) * 16;
    const u32x4* src = (const u32x4*)(Qbf + headBase + (size_t)(q0 + r) * DH + c0);
    *(u32x4*)(Qs + r * 72 + c0)     = src[0];
    *(u32x4*)(Qs + r * 72 + c0 + 8) = src[1];
  }
  if (tid < 64) { mrow[tid] = -1e30f; lrow[tid] = 0.0f; }
  CFrag o[2];
  for (int j = 0; j < 2; j++)
    for (int r = 0; r < 8; r++) o[j].f[r] = 0.0f;
  __syncthreads();

  for (int kc = 0; kc < 8; kc++) {
    const int k0 = start + kc * 64;
    { // load K row-major, V transposed
      int r = tid >> 2, c0 = (tid & 3) * 16;
      const u32x4* ksrc = (const u32x4*)(Kbf + headBase + (size_t)(k0 + r) * DH + c0);
      *(u32x4*)(Ks + r * 72 + c0)     = ksrc[0];
      *(u32x4*)(Ks + r * 72 + c0 + 8) = ksrc[1];
      Pack16 vv;
      const u32x4* vsrc = (const u32x4*)(Vbf + headBase + (size_t)(k0 + r) * DH + c0);
      vv.u[0] = vsrc[0]; vv.u[1] = vsrc[1];
      for (int jj = 0; jj < 16; jj++) Vt[(c0 + jj) * 72 + r] = vv.s[jj];
    }
    __syncthreads();

    { // scores: S = Q K^T * scale + pos_bias  (batch ds_loads, then WMMA chain)
      v16bf qa[2], kf[2][2];
      for (int t = 0; t < 2; t++) {
        qa[t] = load_a_frag(Qs, qr * 16, t * 32, 72, half, ln);
        for (int j = 0; j < 2; j++)
          kf[t][j] = load_b_frag(Ks, (kg * 2 + j) * 16, t * 32, 72, half, ln);
      }
      CFrag sc[2];
      for (int j = 0; j < 2; j++)
        for (int r = 0; r < 8; r++) sc[j].f[r] = 0.0f;
      for (int t = 0; t < 2; t++)
        for (int j = 0; j < 2; j++)
          sc[j].v = wmma_bf16(qa[t], kf[t][j], sc[j].v);

      for (int j = 0; j < 2; j++) {
        int klocal = (kg * 2 + j) * 16 + ln;
        int kw = kc * 64 + klocal;
        for (int r = 0; r < 8; r++) {
          int qlocal = qr * 16 + r + 8 * half;
          int qw = qtile * 64 + qlocal;
          Ss[qlocal * 68 + klocal] =
              sc[j].f[r] * scale + pos_bias[((size_t)h * WIN + qw) * WIN + kw];
        }
      }
    }
    __syncthreads();

    // ---- parallel online softmax: 4 threads per row, 16 cols each ----
    {
      const float* srow = Ss + rr * 68 + seg * 16;
      float pm = -1e30f;
      for (int c = 0; c < 16; c++) pm = fmaxf(pm, srow[c]);
      red[rr * 4 + seg] = pm;
    }
    __syncthreads();
    if (seg == 0) {
      float mc = fmaxf(fmaxf(red[rr * 4 + 0], red[rr * 4 + 1]),
                       fmaxf(red[rr * 4 + 2], red[rr * 4 + 3]));
      float mold = mrow[rr];
      float mnew = fmaxf(mold, mc);
      rfac[rr] = __expf(mold - mnew);
      mrow[rr] = mnew;
    }
    __syncthreads();
    {
      float mnew = mrow[rr];
      float ps = 0.0f;
      for (int c = 0; c < 16; c++) {
        float e = __expf(Ss[rr * 68 + seg * 16 + c] - mnew);
        Ps[rr * 72 + seg * 16 + c] = f2bf(e);
        ps += e;
      }
      red[rr * 4 + seg] = ps;
    }
    __syncthreads();
    if (seg == 0) {
      lrow[rr] = lrow[rr] * rfac[rr] +
                 (red[rr * 4 + 0] + red[rr * 4 + 1]) +
                 (red[rr * 4 + 2] + red[rr * 4 + 3]);
    }
    __syncthreads();

    // rescale accumulators, then O += P V (batch ds_loads, then WMMA chain)
    for (int j = 0; j < 2; j++)
      for (int r = 0; r < 8; r++)
        o[j].f[r] *= rfac[qr * 16 + r + 8 * half];
    {
      v16bf pa[2], vf[2][2];
      for (int t = 0; t < 2; t++) {
        pa[t] = load_a_frag(Ps, qr * 16, t * 32, 72, half, ln);
        for (int j = 0; j < 2; j++)
          vf[t][j] = load_b_frag(Vt, kg * 32 + j * 16, t * 32, 72, half, ln);
      }
      for (int t = 0; t < 2; t++)
        for (int j = 0; j < 2; j++)
          o[j].v = wmma_bf16(pa[t], vf[t][j], o[j].v);
    }
    __syncthreads();
  }

  // epilogue: normalize by l, atomic-merge (overlapping windows)
  for (int j = 0; j < 2; j++) {
    int d = kg * 32 + j * 16 + ln;
    for (int r = 0; r < 8; r++) {
      int qlocal = qr * 16 + r + 8 * half;
      float val = o[j].f[r] / lrow[qlocal];
      int sg = q0 + qlocal;
      atomicAdd(&merged[headBase + (size_t)sg * DH + d], val);
    }
  }
}

// ------------------------------- launcher -----------------------------------

extern "C" void kernel_launch(void* const* d_in, const int* in_sizes, int n_in,
                              void* d_out, int out_size, void* d_ws, size_t ws_size,
                              hipStream_t stream) {
  (void)in_sizes; (void)n_in; (void)out_size; (void)ws_size;
  const float* X   = (const float*)d_in[0];
  const float* Wq  = (const float*)d_in[1];
  const float* bq  = (const float*)d_in[2];
  const float* Wk  = (const float*)d_in[3];
  const float* bk  = (const float*)d_in[4];
  const float* Wv  = (const float*)d_in[5];
  const float* bv  = (const float*)d_in[6];
  const float* Wo  = (const float*)d_in[7];
  const float* bo  = (const float*)d_in[8];
  const float* pos = (const float*)d_in[9];

  char* ws = (char*)d_ws;
  size_t off = 0;
  auto alloc = [&](size_t bytes) -> void* {
    void* p = ws + off;
    off += (bytes + 255) & ~(size_t)255;
    return p;
  };
  const size_t NE = (size_t)Bsz * Ssz * HID;            // 8,192,000
  unsigned short* Xbf  = (unsigned short*)alloc(NE * 2);
  unsigned short* Wt3  = (unsigned short*)alloc((size_t)3 * HID * HID * 2); // qkv^T
  unsigned short* Wto  = (unsigned short*)alloc((size_t)HID * HID * 2);
  unsigned short* Qbf  = (unsigned short*)alloc(NE * 2);
  unsigned short* Kbf  = (unsigned short*)alloc(NE * 2);
  unsigned short* Vbf  = (unsigned short*)alloc(NE * 2);
  float*          merged = (float*)alloc(NE * 4);
  unsigned short* ctx  = (unsigned short*)alloc(NE * 2);

  const int TB = 256;
  const int nElem = (int)NE;
  const int gElem = (nElem + TB - 1) / TB;
  const int gW = (HID * HID + TB - 1) / TB;

  swa_cvt_bf16_kernel<<<gElem, TB, 0, stream>>>(X, Xbf, nElem);
  swa_wtrans_kernel<<<gW, TB, 0, stream>>>(Wq, Wt3);
  swa_wtrans_kernel<<<gW, TB, 0, stream>>>(Wk, Wt3 + (size_t)HID * HID);
  swa_wtrans_kernel<<<gW, TB, 0, stream>>>(Wv, Wt3 + (size_t)2 * HID * HID);
  swa_wtrans_kernel<<<gW, TB, 0, stream>>>(Wo, Wto);
  swa_zero_kernel<<<gElem, TB, 0, stream>>>(merged, nElem);

  dim3 gQKV((MROWS + 127) / 128, (3 * HID) / 128);       // 63 x 24
  swa_gemm_qkv_kernel<<<gQKV, TB, 0, stream>>>(Xbf, Wt3, bq, bk, bv,
                                               Qbf, Kbf, Vbf, MROWS);

  const int nAttnBlocks = Bsz * NH * NWIN * 8;           // 3840
  swa_attn_kernel<<<nAttnBlocks, TB, 0, stream>>>(Qbf, Kbf, Vbf, pos, merged);

  swa_norm_kernel<<<gElem, TB, 0, stream>>>(merged, ctx);

  dim3 gOut((MROWS + 127) / 128, HID / 128);             // 63 x 8
  swa_gemm_out_kernel<<<gOut, TB, 0, stream>>>(ctx, Wto, bo, (float*)d_out, MROWS);
}
